// QPMHALayer_35098472742918
// MI455X (gfx1250) — matrix operand order repaired and checked by reading, not compiled, not run on hardware
//
#include <hip/hip_runtime.h>
#include <hip/hip_bf16.h>

typedef __attribute__((ext_vector_type(16))) _Float16 v16h;
typedef __attribute__((ext_vector_type(8)))  _Float16 v8h;
typedef __attribute__((ext_vector_type(8)))  float    v8f;

#define SCALE_C 0.125f   // 1/sqrt(64)

// ---------------- WMMA helpers (CDNA5 v_wmma_f32_16x16x32_f16) ----------------

__device__ __forceinline__ v8f wmma_f16(v16h a, v16h b, v8f c) {
    return __builtin_amdgcn_wmma_f32_16x16x32_f16(
        /*neg_a=*/false, a, /*neg_b=*/false, b,
        /*c_mod=*/(short)0, c, /*reuse_a=*/false, /*reuse_b=*/false);
}

// A fragment: 16x32 f16 tile from row-major matrix (lda = K stride).
// Lane l: M = row0 + (l&15); K = k0 + (l>>4)*16 + {0..15} (contiguous).
__device__ __forceinline__ v16h load_a_frag(const _Float16* base, int lda, int row0, int k0) {
    int lane = threadIdx.x & 31;
    int m    = row0 + (lane & 15);
    int klo  = (lane >> 4) << 4;
    const _Float16* p = base + (size_t)m * lda + k0 + klo;
    v8h lo = *(const v8h*)(p);
    v8h hi = *(const v8h*)(p + 8);
    return __builtin_shufflevector(lo, hi, 0,1,2,3,4,5,6,7,8,9,10,11,12,13,14,15);
}

// B fragment: 32x16 f16 tile (K x N) from an "n-major" matrix W[n][k] (ldb = K stride).
// Lane l: N = n0 + (l&15); K = k0 + (l>>4)*16 + {0..15} (contiguous).
__device__ __forceinline__ v16h load_b_frag(const _Float16* base, int ldb, int k0, int n0) {
    int lane = threadIdx.x & 31;
    int n    = n0 + (lane & 15);
    int k    = k0 + ((lane >> 4) << 4);
    return *(const v16h*)(base + (size_t)n * ldb + k);
}

// ---------------- Prep kernels ----------------

__global__ void __launch_bounds__(256) cvt_f16_kernel(const float* __restrict__ s,
                                                      _Float16* __restrict__ d, int n) {
    int i = blockIdx.x * 256 + threadIdx.x;
    if (i < n) d[i] = (_Float16)s[i];
}

// d[n*K + k] = (f16) s[k*Nn + n]   (transpose K x Nn -> Nn x K, convert to f16)
__global__ void __launch_bounds__(256) transpose_cvt_kernel(const float* __restrict__ s,
                                                            _Float16* __restrict__ d,
                                                            int K, int Nn) {
    int i = blockIdx.x * 256 + threadIdx.x;
    if (i < K * Nn) {
        int n = i / K;
        int k = i - n * K;
        d[i] = (_Float16)s[(size_t)k * Nn + n];
    }
}

// ---------------- QKV projection: [8192,512] @ [512,1536] + b ----------------
// One wave per 32x64 output tile (2 A-frags x 4 B-frags -> 8 WMMAs per k-step).
// Writes Q (pre-scaled), K as [bh][n][hd], V transposed as [bh][hd][n], all f16.

__global__ void __launch_bounds__(256) qkv_gemm_kernel(const _Float16* __restrict__ xh,
                                                       const _Float16* __restrict__ wT,
                                                       const float* __restrict__ bias,
                                                       _Float16* __restrict__ Qb,
                                                       _Float16* __restrict__ Kb,
                                                       _Float16* __restrict__ Vt) {
    int lane = threadIdx.x & 31;
    int wid  = blockIdx.x * 8 + (threadIdx.x >> 5);  // 0..6143
    int tm   = wid & 255;                            // 256 row tiles of 32 (M=8192)
    int tn   = wid >> 8;                             // 24 col tiles of 64 (N=1536)
    int row0 = tm << 5, col0 = tn << 6;

    v8f acc[2][4] = {};
    for (int k0 = 0; k0 < 512; k0 += 32) {
        v16h a0 = load_a_frag(xh, 512, row0,      k0);
        v16h a1 = load_a_frag(xh, 512, row0 + 16, k0);
        v16h b0 = load_b_frag(wT, 512, k0, col0);
        v16h b1 = load_b_frag(wT, 512, k0, col0 + 16);
        v16h b2 = load_b_frag(wT, 512, k0, col0 + 32);
        v16h b3 = load_b_frag(wT, 512, k0, col0 + 48);
        acc[0][0] = wmma_f16(a0, b0, acc[0][0]);
        acc[0][1] = wmma_f16(a0, b1, acc[0][1]);
        acc[0][2] = wmma_f16(a0, b2, acc[0][2]);
        acc[0][3] = wmma_f16(a0, b3, acc[0][3]);
        acc[1][0] = wmma_f16(a1, b0, acc[1][0]);
        acc[1][1] = wmma_f16(a1, b1, acc[1][1]);
        acc[1][2] = wmma_f16(a1, b2, acc[1][2]);
        acc[1][3] = wmma_f16(a1, b3, acc[1][3]);
    }

#pragma unroll
    for (int bc = 0; bc < 4; bc++) {
        int n     = col0 + bc * 16 + (lane & 15);
        float bn  = bias[n];
        int which = n >> 9;        // 0=q 1=k 2=v (uniform per 64-wide tile)
        int d     = n & 511;
        int h     = d >> 6, hd = d & 63;
#pragma unroll
        for (int ar = 0; ar < 2; ar++) {
#pragma unroll
            for (int r = 0; r < 8; r++) {
                int m  = row0 + ar * 16 + r + ((lane >> 4) << 3);
                int bi = m >> 10, nq = m & 1023;
                int bh = bi * 8 + h;
                float v = acc[ar][bc][r] + bn;
                if (which == 0)
                    Qb[((size_t)bh * 1024 + nq) * 64 + hd] = (_Float16)(v * SCALE_C);
                else if (which == 1)
                    Kb[((size_t)bh * 1024 + nq) * 64 + hd] = (_Float16)v;
                else
                    Vt[((size_t)bh * 64 + hd) * 1024 + nq] = (_Float16)v;
            }
        }
    }
}

// ---------------- Flash attention: one wave per (b,h,16-query tile) ----------------

__global__ void __launch_bounds__(256) attn_kernel(const _Float16* __restrict__ Qb,
                                                   const _Float16* __restrict__ Kb,
                                                   const _Float16* __restrict__ Vt,
                                                   const float* __restrict__ U,
                                                   const float* __restrict__ mask,
                                                   _Float16* __restrict__ ctx) {
    __shared__ __align__(32) _Float16 pbuf[8][16 * 32];  // per-wave P staging (C->A layout)

    int lane = threadIdx.x & 31;
    int warp = threadIdx.x >> 5;
    int wid  = blockIdx.x * 8 + warp;   // 0..4095
    int bh   = wid >> 6;                // 0..63
    int qt   = wid & 63;
    int b    = bh >> 3;
    int h    = bh & 7;
    int q0   = qt << 4;

    const _Float16* Qp = Qb + (size_t)bh * (1024 * 64);
    const _Float16* Kp = Kb + (size_t)bh * (1024 * 64);
    const _Float16* Vp = Vt + (size_t)bh * (64 * 1024);
    const float*    mp = mask + b * 1024;
    _Float16*       pw = &pbuf[warp][0];

    // Q A-fragments (held for the whole key loop), K=64 in two chunks
    v16h aq0 = load_a_frag(Qp, 64, q0, 0);
    v16h aq1 = load_a_frag(Qp, 64, q0, 32);

    v8f acc0 = {}, acc1 = {}, acc2 = {}, acc3 = {};
    float mi[8], li[8];
    int qrow[8];
#pragma unroll
    for (int r = 0; r < 8; r++) {
        mi[r] = -3.0e38f; li[r] = 0.0f;
        qrow[r] = q0 + r + ((lane >> 4) << 3);
    }

    for (int kt = 0; kt < 1024; kt += 32) {
        // scores 16x32 = Q(16x64) @ K^T(64x32), two 16x16 C tiles
        v8f s0 = {}, s1 = {};
        s0 = wmma_f16(aq0, load_b_frag(Kp, 64, 0,  kt),      s0);
        s0 = wmma_f16(aq1, load_b_frag(Kp, 64, 32, kt),      s0);
        s1 = wmma_f16(aq0, load_b_frag(Kp, 64, 0,  kt + 16), s1);
        s1 = wmma_f16(aq1, load_b_frag(Kp, 64, 32, kt + 16), s1);

        int key0 = kt + (lane & 15);
        int key1 = key0 + 16;
        float mb0 = mp[key0], mb1 = mp[key1];
#pragma unroll
        for (int r = 0; r < 8; r++) {
            s0[r] += U[(size_t)qrow[r] * 1024 + key0] + mb0;
            s1[r] += U[(size_t)qrow[r] * 1024 + key1] + mb1;
        }

        // row max across the 16-lane half (rows identical across those lanes)
        float mt[8];
#pragma unroll
        for (int r = 0; r < 8; r++) mt[r] = fmaxf(s0[r], s1[r]);
#pragma unroll
        for (int off = 1; off < 16; off <<= 1) {
#pragma unroll
            for (int r = 0; r < 8; r++)
                mt[r] = fmaxf(mt[r], __shfl_xor(mt[r], off, 32));
        }

        float fac[8];
#pragma unroll
        for (int r = 0; r < 8; r++) {
            float mn = fmaxf(mi[r], mt[r]);
            fac[r] = __expf(mi[r] - mn);
            mi[r] = mn;
        }

        // P = exp(S - m), stash f16 P tile in LDS (row-major 16x32) for A-frag reload
        float rs[8];
#pragma unroll
        for (int r = 0; r < 8; r++) {
            float p0 = __expf(s0[r] - mi[r]);
            float p1 = __expf(s1[r] - mi[r]);
            rs[r] = p0 + p1;
            int m = r + ((lane >> 4) << 3);
            pw[m * 32 + (lane & 15)]      = (_Float16)p0;
            pw[m * 32 + 16 + (lane & 15)] = (_Float16)p1;
        }
#pragma unroll
        for (int off = 1; off < 16; off <<= 1) {
#pragma unroll
            for (int r = 0; r < 8; r++)
                rs[r] += __shfl_xor(rs[r], off, 32);
        }
#pragma unroll
        for (int r = 0; r < 8; r++) {
            li[r] = li[r] * fac[r] + rs[r];
            acc0[r] *= fac[r]; acc1[r] *= fac[r];
            acc2[r] *= fac[r]; acc3[r] *= fac[r];
        }

        // ctx += P(16x32) @ V(32x64): Vt is [hd][key] so B-frags are contiguous
        v16h ap = load_a_frag(pw, 32, 0, 0);
        acc0 = wmma_f16(ap, load_b_frag(Vp, 1024, kt, 0),  acc0);
        acc1 = wmma_f16(ap, load_b_frag(Vp, 1024, kt, 16), acc1);
        acc2 = wmma_f16(ap, load_b_frag(Vp, 1024, kt, 32), acc2);
        acc3 = wmma_f16(ap, load_b_frag(Vp, 1024, kt, 48), acc3);
    }

    // write ctx as [B, N, H, HD] f16 (== merged-head [8192, 512])
#pragma unroll
    for (int r = 0; r < 8; r++) {
        float inv = 1.0f / li[r];
        size_t rowb = (((size_t)b * 1024 + qrow[r]) * 8 + h) * 64;
        int c = lane & 15;
        ctx[rowb + 0  + c] = (_Float16)(acc0[r] * inv);
        ctx[rowb + 16 + c] = (_Float16)(acc1[r] * inv);
        ctx[rowb + 32 + c] = (_Float16)(acc2[r] * inv);
        ctx[rowb + 48 + c] = (_Float16)(acc3[r] * inv);
    }
}

// ---------------- Output projection: [8192,512] @ [512,512] + b -> f32 ----------------
// One wave per 32x64 output tile.

__global__ void __launch_bounds__(256) out_gemm_kernel(const _Float16* __restrict__ ctx,
                                                       const _Float16* __restrict__ wT,
                                                       const float* __restrict__ bias,
                                                       float* __restrict__ out) {
    int lane = threadIdx.x & 31;
    int wid  = blockIdx.x * 8 + (threadIdx.x >> 5);  // 0..2047
    int tm   = wid & 255;                            // 256 row tiles of 32
    int tn   = wid >> 8;                             // 8 col tiles of 64
    int row0 = tm << 5, col0 = tn << 6;

    v8f acc[2][4] = {};
    for (int k0 = 0; k0 < 512; k0 += 32) {
        v16h a0 = load_a_frag(ctx, 512, row0,      k0);
        v16h a1 = load_a_frag(ctx, 512, row0 + 16, k0);
        v16h b0 = load_b_frag(wT, 512, k0, col0);
        v16h b1 = load_b_frag(wT, 512, k0, col0 + 16);
        v16h b2 = load_b_frag(wT, 512, k0, col0 + 32);
        v16h b3 = load_b_frag(wT, 512, k0, col0 + 48);
        acc[0][0] = wmma_f16(a0, b0, acc[0][0]);
        acc[0][1] = wmma_f16(a0, b1, acc[0][1]);
        acc[0][2] = wmma_f16(a0, b2, acc[0][2]);
        acc[0][3] = wmma_f16(a0, b3, acc[0][3]);
        acc[1][0] = wmma_f16(a1, b0, acc[1][0]);
        acc[1][1] = wmma_f16(a1, b1, acc[1][1]);
        acc[1][2] = wmma_f16(a1, b2, acc[1][2]);
        acc[1][3] = wmma_f16(a1, b3, acc[1][3]);
    }

#pragma unroll
    for (int bc = 0; bc < 4; bc++) {
        int n    = col0 + bc * 16 + (lane & 15);
        float bn = bias[n];
#pragma unroll
        for (int ar = 0; ar < 2; ar++) {
#pragma unroll
            for (int r = 0; r < 8; r++) {
                int m = row0 + ar * 16 + r + ((lane >> 4) << 3);
                out[(size_t)m * 512 + n] = acc[ar][bc][r] + bn;
            }
        }
    }
}

// ---------------- Host launcher ----------------

extern "C" void kernel_launch(void* const* d_in, const int* in_sizes, int n_in,
                              void* d_out, int out_size, void* d_ws, size_t ws_size,
                              hipStream_t stream) {
    (void)in_sizes; (void)n_in; (void)out_size; (void)ws_size;
    const float* x      = (const float*)d_in[0];  // [8,1024,512]
    const float* U      = (const float*)d_in[1];  // [1024,1024]
    const float* mask   = (const float*)d_in[2];  // [8,1,1,1024]
    const float* w_qkv  = (const float*)d_in[3];  // [512,1536]
    const float* b_qkv  = (const float*)d_in[4];  // [1536]
    const float* w_out  = (const float*)d_in[5];  // [512,512]
    const float* b_out  = (const float*)d_in[6];  // [512]

    char* w = (char*)d_ws;
    _Float16* xh   = (_Float16*)(w + 0);            //  8,388,608 B  x in f16
    _Float16* wqT  = (_Float16*)(w + 8388608);      //  1,572,864 B  w_qkv^T f16 [1536,512]
    _Float16* woT  = (_Float16*)(w + 10485760);     //    524,288 B  w_out^T f16 [512,512]
    _Float16* Qb   = (_Float16*)(w + 11534336);     //  8,388,608 B  [bh][n][hd]
    _Float16* Kb   = (_Float16*)(w + 19922944);     //  8,388,608 B  [bh][n][hd]
    _Float16* Vt   = (_Float16*)(w + 28311552);     //  8,388,608 B  [bh][hd][n]
    _Float16* ctxh = (_Float16*)(w + 36700160);     //  8,388,608 B  [8192,512]

    // prep: fp32 -> f16 conversions / weight transposes
    cvt_f16_kernel<<<(8192 * 512 + 255) / 256, 256, 0, stream>>>(x, xh, 8192 * 512);
    transpose_cvt_kernel<<<(512 * 1536 + 255) / 256, 256, 0, stream>>>(w_qkv, wqT, 512, 1536);
    transpose_cvt_kernel<<<(512 * 512 + 255) / 256, 256, 0, stream>>>(w_out, woT, 512, 512);

    // QKV projection: 256x24 wave-tiles (32x64 each), 8 waves/block
    qkv_gemm_kernel<<<768, 256, 0, stream>>>(xh, wqT, b_qkv, Qb, Kb, Vt);

    // flash attention: 64 (b,h) x 64 q-tiles = 4096 waves
    attn_kernel<<<512, 256, 0, stream>>>(Qb, Kb, Vt, U, mask, ctxh);

    // output projection -> fp32 d_out
    out_gemm_kernel<<<256, 256, 0, stream>>>(ctxh, woT, b_out, (float*)d_out);
}
